// QuantizedVisionLanguageAttentionKernel_89223650607646
// MI455X (gfx1250) — matrix operand order repaired and checked by reading, not compiled, not run on hardware
//
#include <hip/hip_runtime.h>
#include <hip/hip_bf16.h>
#include <stdint.h>

#define D_    2048
#define H_    16
#define HD_   128
#define B_    4
#define P_    576
#define S_    1024
#define NVROWS (B_*P_)        // 2304
#define NLROWS (B_*S_)        // 4096
#define NFROWS (B_*(P_+S_))   // 6400

typedef __bf16 bf16;
typedef __attribute__((ext_vector_type(16))) __bf16 v16bf;
typedef __attribute__((ext_vector_type(8)))  float  v8f;
typedef unsigned int u32x4 __attribute__((ext_vector_type(4)));
typedef int i32x4 __attribute__((ext_vector_type(4)));
typedef int i32x8 __attribute__((ext_vector_type(8)));

#if defined(__has_builtin)
#if __has_builtin(__builtin_amdgcn_tensor_load_to_lds)
#define HAS_TDM 1
#endif
#endif

union Frag16 { uint4 q[2]; v16bf v; };

// A fragment: 16x32 bf16 (MxK), row-major, row stride lda (elements).
// lane: M = lane%16, half = lane/16; K = 16*h .. 16*h+15 contiguous (32 bytes).
__device__ __forceinline__ v16bf load_a_frag(const bf16* base, int lda, int lane) {
    int row = lane & 15, h = lane >> 4;
    const bf16* p = base + (size_t)row * lda + h * 16;
    Frag16 f;
    f.q[0] = *reinterpret_cast<const uint4*>(p);
    f.q[1] = *reinterpret_cast<const uint4*>(p + 8);
    return f.v;
}

// B fragment: 32x16 bf16 (KxN). Element (k,n) = base[n*ldb + k] (n-major rows).
// lane: N = lane%16, half = lane/16; K = 16*h .. 16*h+15 contiguous (32 bytes).
__device__ __forceinline__ v16bf load_b_frag(const bf16* base, int ldb, int lane) {
    int col = lane & 15, h = lane >> 4;
    const bf16* p = base + (size_t)col * ldb + h * 16;
    Frag16 f;
    f.q[0] = *reinterpret_cast<const uint4*>(p);
    f.q[1] = *reinterpret_cast<const uint4*>(p + 8);
    return f.v;
}

__device__ __forceinline__ v8f wmma_bf16(v16bf a, v16bf b, v8f c) {
    return __builtin_amdgcn_wmma_f32_16x16x32_bf16(false, a, false, b, (short)0, c,
                                                   false, false);
}

#ifdef HAS_TDM
// TDM: DMA a 16-row x 1024-col bf16 tile of W (row stride 2048 elems) into LDS.
// Descriptor packing per CDNA5 ISA 8.3/8.4 (2D tensor, groups 2/3 unused).
__device__ __forceinline__ void tdm_load_w_tile(const bf16* gsrc, bf16* ldsdst) {
    unsigned lds_off = (unsigned)(size_t)(void*)ldsdst;  // low 32b = LDS offset
    unsigned long long ga = (unsigned long long)(size_t)gsrc;
    u32x4 g0;
    g0.x = 1u;                                            // count=1, user mode
    g0.y = lds_off;                                       // lds_addr [63:32]
    g0.z = (unsigned)(ga & 0xffffffffu);                  // global_addr lo
    g0.w = (unsigned)((ga >> 32) & 0x01ffffffu) | (2u << 30);  // addr hi | type=2
    i32x8 g1;
    g1[0] = (int)(1u << 16);       // workgroup_mask=0 | data_size=1 (2 bytes)
    g1[1] = (int)(2048u << 16);    // tensor_dim0[15:0]=2048 (bits 79:48 lo)
    g1[2] = (int)(2048u << 16);    // tensor_dim0 hi=0 | tensor_dim1 lo=2048
    g1[3] = (int)(1024u << 16);    // tensor_dim1 hi=0 | tile_dim0=1024
    g1[4] = 16;                    // tile_dim1=16 | tile_dim2=0
    g1[5] = 2048;                  // tensor_dim0_stride lo = 2048 elems
    g1[6] = 0;                     // stride hi | tensor_dim1_stride lo
    g1[7] = 0;
    i32x4 z4 = {0, 0, 0, 0};
#if __clang_major__ >= 23
    i32x8 z8 = {0, 0, 0, 0, 0, 0, 0, 0};
    __builtin_amdgcn_tensor_load_to_lds(g0, g1, z4, z4, z8, 0);
#else
    __builtin_amdgcn_tensor_load_to_lds(g0, g1, z4, z4, 0);
#endif
}
#endif

// B fragment from the LDS-staged W tile (row stride 1024 elements).
__device__ __forceinline__ v16bf load_b_frag_lds(const bf16* buf, int kk, int lane) {
    int col = lane & 15, h = lane >> 4;
    const bf16* p = buf + col * 1024 + kk + h * 16;
    Frag16 f;
    f.q[0] = *reinterpret_cast<const uint4*>(p);
    f.q[1] = *reinterpret_cast<const uint4*>(p + 8);
    return f.v;
}

// ---------------------------------------------------------------------------
// 1) pos-add + LayerNorm: fp32 residual + bf16 normalized
// ---------------------------------------------------------------------------
__global__ void prep_ln_kernel(const float* __restrict__ visf, const float* __restrict__ langf,
                               const float* __restrict__ vpos, const float* __restrict__ tpos,
                               const float* __restrict__ vg, const float* __restrict__ vb,
                               const float* __restrict__ lg, const float* __restrict__ lb,
                               float* __restrict__ vis_res, float* __restrict__ lang_res,
                               bf16* __restrict__ visn, bf16* __restrict__ langn) {
    int row = blockIdx.x;
    const float *xin, *pos, *g, *bb;
    float* res; bf16* nout;
    if (row < NVROWS) {
        int p = row % P_;
        xin = visf + (size_t)row * D_;  pos = vpos + (size_t)p * D_;
        g = vg; bb = vb;
        res = vis_res + (size_t)row * D_;  nout = visn + (size_t)row * D_;
    } else {
        int j = row - NVROWS; int s = j % S_;
        xin = langf + (size_t)j * D_;  pos = tpos + (size_t)s * D_;
        g = lg; bb = lb;
        res = lang_res + (size_t)j * D_;  nout = langn + (size_t)j * D_;
    }
    int tid = threadIdx.x;
    float x[8]; float s1 = 0.f, s2 = 0.f;
#pragma unroll
    for (int i = 0; i < 8; ++i) {
        int e = tid + i * 256;
        float v = xin[e] + pos[e];
        x[i] = v; s1 += v; s2 += v * v;
    }
    __shared__ float r1[256], r2[256];
    r1[tid] = s1; r2[tid] = s2; __syncthreads();
    for (int st = 128; st > 0; st >>= 1) {
        if (tid < st) { r1[tid] += r1[tid + st]; r2[tid] += r2[tid + st]; }
        __syncthreads();
    }
    float mean = r1[0] * (1.0f / D_);
    float var  = r2[0] * (1.0f / D_) - mean * mean;
    float rstd = rsqrtf(var + 1e-5f);
#pragma unroll
    for (int i = 0; i < 8; ++i) {
        int e = tid + i * 256;
        res[e]  = x[i];
        nout[e] = (bf16)((x[i] - mean) * rstd * g[e] + bb[e]);
    }
}

// ---------------------------------------------------------------------------
// 2) fp32 -> bf16 convert
// ---------------------------------------------------------------------------
__global__ void f2bf_kernel(const float* __restrict__ in, bf16* __restrict__ out, int n) {
    int i = blockIdx.x * blockDim.x + threadIdx.x;
    int stride = gridDim.x * blockDim.x;
    for (; i < n; i += stride) out[i] = (bf16)in[i];
}

// ---------------------------------------------------------------------------
// 3) C[M,2048] = alpha*(A[M,2048] @ W^T + bias) (+ resid); W row-major [N,K].
//    Block: 1 N-tile x 8 M-tiles (8 waves). The shared 16x2048 W strip is
//    staged into LDS by the Tensor Data Mover in two 32KB halves
//    (double-buffered: second DMA overlaps first half's WMMA loop).
// ---------------------------------------------------------------------------
__global__ void gemm_xwT_kernel(const bf16* __restrict__ A, const bf16* __restrict__ W,
                                const float* __restrict__ bias, float alpha,
                                const float* __restrict__ resid,
                                float* __restrict__ outF, bf16* __restrict__ outB,
                                int chunk, int outerStride, int rowOff) {
    int lane = threadIdx.x & 31;
    int wave = threadIdx.x >> 5;
    int tn = blockIdx.x & 127;            // 2048/16 = 128 N tiles
    int bm = blockIdx.x >> 7;
    int tm = bm * 8 + wave;

    __shared__ __align__(16) bf16 ldsW[2][16 * 1024];   // 64 KB
    const bf16* Wtile = W + (size_t)(tn * 16) * D_;

#ifdef HAS_TDM
    if (wave == 0) {
        tdm_load_w_tile(Wtile, &ldsW[0][0]);            // half 0: k = 0..1023
        __builtin_amdgcn_s_wait_tensorcnt(0);
    }
    __syncthreads();
    if (wave == 0) {
        tdm_load_w_tile(Wtile + 1024, &ldsW[1][0]);     // half 1 overlaps compute
    }
#else
    for (int half = 0; half < 2; ++half)
        for (int i = threadIdx.x; i < 16 * 1024; i += 256) {
            int r = i >> 10, c = i & 1023;
            ldsW[half][i] = Wtile[(size_t)r * D_ + half * 1024 + c];
        }
    __syncthreads();
#endif

    const bf16* Abase = A + (size_t)(tm * 16) * D_;
    v8f acc = {};
#pragma unroll 4
    for (int kk = 0; kk < 1024; kk += 32) {
        __builtin_prefetch(Abase + kk + 1024, 0, 0);    // global_prefetch_b8
        v16bf a = load_a_frag(Abase + kk, D_, lane);
        v16bf b = load_b_frag_lds(&ldsW[0][0], kk, lane);
        acc = wmma_bf16(a, b, acc);
    }
#ifdef HAS_TDM
    if (wave == 0) __builtin_amdgcn_s_wait_tensorcnt(0);
#endif
    __syncthreads();
#pragma unroll 4
    for (int kk = 0; kk < 1024; kk += 32) {
        v16bf a = load_a_frag(Abase + 1024 + kk, D_, lane);
        v16bf b = load_b_frag_lds(&ldsW[1][0], kk, lane);
        acc = wmma_bf16(a, b, acc);
    }

    int h = lane >> 4;
    int n = tn * 16 + (lane & 15);
    float bn = bias[n];
#pragma unroll
    for (int r = 0; r < 8; ++r) {
        int m = tm * 16 + h * 8 + r;
        float v = alpha * (acc[r] + bn);
        if (resid) v += resid[(size_t)m * D_ + n];
        if (outF)  outF[(size_t)m * D_ + n] = v;
        if (outB) {
            int mrow = (m / chunk) * outerStride + rowOff + (m % chunk);
            outB[(size_t)mrow * D_ + n] = (bf16)v;
        }
    }
}

// ---------------------------------------------------------------------------
// 4) scores[b,h] = Q[b,:,h*128..] @ K[b,:,h*128..]^T   (fp32 out)
// ---------------------------------------------------------------------------
__global__ void attn_scores_kernel(const bf16* __restrict__ Q, const bf16* __restrict__ Km,
                                   float* __restrict__ S, int Lq, int Lk) {
    int gy = blockIdx.y;                  // b*H + h
    int b = gy >> 4, h = gy & 15;
    int lane = threadIdx.x & 31, wave = threadIdx.x >> 5;
    int tile = blockIdx.x * 8 + wave;
    int nkt = Lk >> 4;
    int tn = tile % nkt, tm = tile / nkt;
    const bf16* Qb = Q + ((size_t)b * Lq + tm * 16) * D_ + h * HD_;
    const bf16* Kb = Km + ((size_t)b * Lk + tn * 16) * D_ + h * HD_;
    float* Sb = S + (size_t)gy * Lq * Lk;
    v8f acc = {};
#pragma unroll
    for (int kk = 0; kk < HD_; kk += 32) {
        v16bf a = load_a_frag(Qb + kk, D_, lane);
        v16bf bb = load_b_frag(Kb + kk, D_, lane);
        acc = wmma_bf16(a, bb, acc);
    }
    int hh = lane >> 4, n = tn * 16 + (lane & 15);
#pragma unroll
    for (int r = 0; r < 8; ++r) {
        int m = tm * 16 + hh * 8 + r;
        Sb[(size_t)m * Lk + n] = acc[r];
    }
}

// ---------------------------------------------------------------------------
// 5) row softmax: fp32 out (may alias input) + bf16 out
// ---------------------------------------------------------------------------
__global__ void softmax_kernel(const float* __restrict__ Sin, float* __restrict__ Fout,
                               bf16* __restrict__ Bout, int Lk) {
    size_t row = blockIdx.x;
    const float* x = Sin + row * Lk;
    float* fo = Fout + row * Lk;
    bf16*  bo = Bout + row * Lk;
    int tid = threadIdx.x;
    __shared__ float red[256];
    float mx = -3.4e38f;
    for (int i = tid; i < Lk; i += 256) mx = fmaxf(mx, x[i]);
    red[tid] = mx; __syncthreads();
    for (int st = 128; st > 0; st >>= 1) {
        if (tid < st) red[tid] = fmaxf(red[tid], red[tid + st]);
        __syncthreads();
    }
    mx = red[0]; __syncthreads();
    float sum = 0.f;
    for (int i = tid; i < Lk; i += 256) sum += __expf(x[i] - mx);
    red[tid] = sum; __syncthreads();
    for (int st = 128; st > 0; st >>= 1) {
        if (tid < st) red[tid] += red[tid + st];
        __syncthreads();
    }
    float inv = 1.f / red[0];
    for (int i = tid; i < Lk; i += 256) {
        float p = __expf(x[i] - mx) * inv;
        fo[i] = p; bo[i] = (bf16)p;
    }
}

// ---------------------------------------------------------------------------
// 6) per-head transpose: V[B,L,D] -> Vt[B*H, 128, L]
// ---------------------------------------------------------------------------
__global__ void vtrans_kernel(const bf16* __restrict__ V, bf16* __restrict__ Vt, int L) {
    size_t total = (size_t)B_ * L * D_;
    size_t i = (size_t)blockIdx.x * blockDim.x + threadIdx.x;
    size_t stride = (size_t)gridDim.x * blockDim.x;
    for (; i < total; i += stride) {
        size_t b = i / ((size_t)L * D_);
        size_t r = i % ((size_t)L * D_);
        int s = (int)(r / D_), c = (int)(r % D_);
        int hh = c >> 7, d = c & 127;
        Vt[(((size_t)b * H_ + hh) * HD_ + d) * L + s] = V[i];
    }
}

// ---------------------------------------------------------------------------
// 7) ctx[b,:,h*128..] = probs[b,h] @ V[b,h]   (bf16 merged-head out)
// ---------------------------------------------------------------------------
__global__ void attn_av_kernel(const bf16* __restrict__ Pm, const bf16* __restrict__ Vt,
                               bf16* __restrict__ Ctx, int Lq, int Lk) {
    int gy = blockIdx.y;
    int b = gy >> 4, h = gy & 15;
    int lane = threadIdx.x & 31, wave = threadIdx.x >> 5;
    int tile = blockIdx.x * 8 + wave;
    int tn = tile & 7;       // 128/16 = 8 tiles over head_dim
    int tm = tile >> 3;
    const bf16* Pb = Pm + (size_t)gy * Lq * Lk + (size_t)(tm * 16) * Lk;
    const bf16* Vb = Vt + ((size_t)gy * HD_ + tn * 16) * Lk;
    v8f acc = {};
    for (int kk = 0; kk < Lk; kk += 32) {
        v16bf a = load_a_frag(Pb + kk, Lk, lane);
        v16bf bb = load_b_frag(Vb + kk, Lk, lane);
        acc = wmma_bf16(a, bb, acc);
    }
    int hh = lane >> 4, d = tn * 16 + (lane & 15);
#pragma unroll
    for (int r = 0; r < 8; ++r) {
        int m = tm * 16 + hh * 8 + r;
        Ctx[((size_t)b * Lq + m) * D_ + h * HD_ + d] = (bf16)acc[r];
    }
}

// ---------------------------------------------------------------------------
extern "C" void kernel_launch(void* const* d_in, const int* in_sizes, int n_in,
                              void* d_out, int out_size, void* d_ws, size_t ws_size,
                              hipStream_t stream) {
    (void)in_sizes; (void)n_in; (void)out_size; (void)ws_size;
    const float* w[9]; const float* bias[9];
    for (int i = 0; i < 9; ++i) {
        w[i]    = (const float*)d_in[2 * i];
        bias[i] = (const float*)d_in[2 * i + 1];
    }
    // order: vq vk vv lq lk lv v2l l2v out
    const float* visf  = (const float*)d_in[18];
    const float* langf = (const float*)d_in[19];
    const float* vn_g  = (const float*)d_in[20];
    const float* vn_b  = (const float*)d_in[21];
    const float* ln_g  = (const float*)d_in[22];
    const float* ln_b  = (const float*)d_in[23];
    const float* vpos  = (const float*)d_in[24];
    const float* tpos  = (const float*)d_in[25];

    float* out       = (float*)d_out;
    float* out_fused = out;                                   // 4*1600*2048
    float* out_vis   = out + (size_t)NFROWS * D_;             // 4*576*2048
    float* out_lang  = out_vis + (size_t)NVROWS * D_;         // 4*1024*2048
    float* out_attn  = out_lang + (size_t)NLROWS * D_;        // 4*16*576*1024

    char* ws = (char*)d_ws;
    size_t off = 0;
    auto alloc = [&](size_t bytes) -> void* {
        void* p = ws + off;
        off += (bytes + 255) & ~(size_t)255;
        return p;
    };
    float* vis_res  = (float*)alloc((size_t)NVROWS * D_ * 4);
    float* lang_res = (float*)alloc((size_t)NLROWS * D_ * 4);
    bf16*  visn     = (bf16*) alloc((size_t)NVROWS * D_ * 2);
    bf16*  langn    = (bf16*) alloc((size_t)NLROWS * D_ * 2);
    bf16*  wb[9];
    for (int i = 0; i < 9; ++i) wb[i] = (bf16*)alloc((size_t)D_ * D_ * 2);
    bf16* q_v = (bf16*)alloc((size_t)NVROWS * D_ * 2);
    bf16* k_v = (bf16*)alloc((size_t)NVROWS * D_ * 2);
    bf16* v_v = (bf16*)alloc((size_t)NVROWS * D_ * 2);
    bf16* q_l = (bf16*)alloc((size_t)NLROWS * D_ * 2);
    bf16* k_l = (bf16*)alloc((size_t)NLROWS * D_ * 2);
    bf16* v_l = (bf16*)alloc((size_t)NLROWS * D_ * 2);
    bf16* vt_v = (bf16*)alloc((size_t)B_ * H_ * HD_ * P_ * 2);
    bf16* vt_l = (bf16*)alloc((size_t)B_ * H_ * HD_ * S_ * 2);
    float* scores = (float*)alloc((size_t)B_ * H_ * P_ * S_ * 4);  // shared both dirs
    bf16*  probs  = (bf16*) alloc((size_t)B_ * H_ * P_ * S_ * 2);  // shared both dirs
    bf16*  ctx_v  = (bf16*) alloc((size_t)NVROWS * D_ * 2);
    bf16*  ctx_l  = (bf16*) alloc((size_t)NLROWS * D_ * 2);
    bf16*  concat = (bf16*) alloc((size_t)NFROWS * D_ * 2);

    const float SCALING = 0.08838834764831845f;  // 128^-0.5

    // 1) pos + layernorm
    prep_ln_kernel<<<NVROWS + NLROWS, 256, 0, stream>>>(
        visf, langf, vpos, tpos, vn_g, vn_b, ln_g, ln_b,
        vis_res, lang_res, visn, langn);

    // 2) weights -> bf16
    for (int i = 0; i < 9; ++i)
        f2bf_kernel<<<4096, 256, 0, stream>>>(w[i], wb[i], D_ * D_);

    // 3) QKV projections (grid = (Mt/8) * 128 N-tiles)
    const int GV = (NVROWS / 16 / 8) * 128;   // 2304 blocks
    const int GL = (NLROWS / 16 / 8) * 128;   // 4096 blocks
    gemm_xwT_kernel<<<GV, 256, 0, stream>>>(visn, wb[0], bias[0], SCALING, nullptr,
                                            nullptr, q_v, NVROWS, NVROWS, 0);
    gemm_xwT_kernel<<<GV, 256, 0, stream>>>(visn, wb[1], bias[1], 1.0f, nullptr,
                                            nullptr, k_v, NVROWS, NVROWS, 0);
    gemm_xwT_kernel<<<GV, 256, 0, stream>>>(visn, wb[2], bias[2], 1.0f, nullptr,
                                            nullptr, v_v, NVROWS, NVROWS, 0);
    gemm_xwT_kernel<<<GL, 256, 0, stream>>>(langn, wb[3], bias[3], SCALING, nullptr,
                                            nullptr, q_l, NLROWS, NLROWS, 0);
    gemm_xwT_kernel<<<GL, 256, 0, stream>>>(langn, wb[4], bias[4], 1.0f, nullptr,
                                            nullptr, k_l, NLROWS, NLROWS, 0);
    gemm_xwT_kernel<<<GL, 256, 0, stream>>>(langn, wb[5], bias[5], 1.0f, nullptr,
                                            nullptr, v_l, NLROWS, NLROWS, 0);

    // 4) per-head transposes of V
    vtrans_kernel<<<4096, 256, 0, stream>>>(v_v, vt_v, P_);
    vtrans_kernel<<<4096, 256, 0, stream>>>(v_l, vt_l, S_);

    // ---- v2l: vision queries attend language keys ----
    attn_scores_kernel<<<dim3(288, 64), 256, 0, stream>>>(q_v, k_l, scores, P_, S_);
    softmax_kernel<<<B_ * H_ * P_, 256, 0, stream>>>(scores, out_attn, probs, S_);
    attn_av_kernel<<<dim3(36, 64), 256, 0, stream>>>(probs, vt_l, ctx_v, P_, S_);
    // v2l out-proj + residual -> vision_output (fp32) + concat rows [b*1600 .. +575]
    gemm_xwT_kernel<<<GV, 256, 0, stream>>>(ctx_v, wb[6], bias[6], 1.0f, vis_res,
                                            out_vis, concat, P_, P_ + S_, 0);

    // ---- l2v: language queries attend vision keys ----
    attn_scores_kernel<<<dim3(288, 64), 256, 0, stream>>>(q_l, k_v, scores, S_, P_);
    softmax_kernel<<<B_ * H_ * S_, 256, 0, stream>>>(scores, scores, probs, P_);
    attn_av_kernel<<<dim3(64, 64), 256, 0, stream>>>(probs, vt_v, ctx_l, S_, P_);
    // l2v out-proj + residual -> language_output (fp32) + concat rows [b*1600+576 ..]
    gemm_xwT_kernel<<<GL, 256, 0, stream>>>(ctx_l, wb[7], bias[7], 1.0f, lang_res,
                                            out_lang, concat, S_, P_ + S_, P_);

    // ---- fused output projection over concat [6400, 2048] ----
    const int GF = (NFROWS / 16 / 8) * 128;   // 6400 blocks
    gemm_xwT_kernel<<<GF, 256, 0, stream>>>(concat, wb[8], bias[8], 1.0f, nullptr,
                                            out_fused, nullptr, NFROWS, NFROWS, 0);
}
